// GAT_L_NO_FIXED_OURPUT_DIM_8564164788537
// MI455X (gfx1250) — matrix operand order repaired and checked by reading, not compiled, not run on hardware
//
#include <hip/hip_runtime.h>
#include <hip/hip_bf16.h>

#define N_NODES 20000
#define E_EDGES 320000
#define ET_EDGES (E_EDGES + N_NODES)
#define G_GRAPHS 64
#define EPS_BN 1e-5f
#define CD(a, b) (((a) + (b) - 1) / (b))

typedef _Float16 v8h  __attribute__((ext_vector_type(8)));
typedef _Float16 v16h __attribute__((ext_vector_type(16)));
typedef float    v8f  __attribute__((ext_vector_type(8)));

static __device__ __forceinline__ float leaky02(float x) { return x >= 0.f ? x : 0.2f * x; }

// monotone uint key for float atomic-max (handles negatives)
static __device__ __forceinline__ unsigned f2key(float f) {
  unsigned v = __float_as_uint(f);
  return (v & 0x80000000u) ? ~v : (v | 0x80000000u);
}
static __device__ __forceinline__ float key2f(unsigned k) {
  unsigned v = (k & 0x80000000u) ? (k ^ 0x80000000u) : ~k;
  return __uint_as_float(v);
}

static __device__ __forceinline__ void edge_sd(const int* __restrict__ ei, int e, int& s, int& d) {
  if (e < E_EDGES) { s = ei[e]; d = ei[E_EDGES + e]; }
  else { s = d = e - E_EDGES; }  // self loops appended
}

// ---------------- utility kernels ----------------
__global__ void fill_zero_kernel(float* p, int n) {
  int i = blockIdx.x * blockDim.x + threadIdx.x;
  if (i < n) p[i] = 0.f;
}

__global__ void cast_f16_kernel(const float* __restrict__ src, _Float16* __restrict__ dst, int n) {
  int i = blockIdx.x * blockDim.x + threadIdx.x;
  if (i < n) dst[i] = (_Float16)src[i];
}

// Repack W[K,Nc] (row-major f32) into WMMA B-operand lane order, f16.
// packed index p = ((ct*KT + kt)*32 + lane)*16 + i
//   lane<16 : N = ct*16+lane,    K = kt*32 + (i<8 ? i : 16+i-8)
//   lane>=16: N = ct*16+lane-16, K = kt*32 + 8 + (i<8 ? i : 16+i-8)
__global__ void pack_b_kernel(const float* __restrict__ W, _Float16* __restrict__ Bp, int K, int Nc) {
  int t = blockIdx.x * blockDim.x + threadIdx.x;
  if (t >= K * Nc) return;
  int KT = K >> 5;
  int i    = t & 15;
  int lane = (t >> 4) & 31;
  int q    = t >> 9;        // ct*KT + kt
  int kt = q % KT;
  int ct = q / KT;
  int n = (ct << 4) + (lane & 15);
  int k = (kt << 5) + ((lane >> 4) << 3) + (i & 7) + ((i >> 3) << 4);
  Bp[t] = (_Float16)W[k * Nc + n];
}

// ---------------- WMMA GEMM: C[M,Nc] = A[M,K](f16) @ Bp(packed f16), fp32 accum ----------------
// grid.x = M/16 row tiles, grid.y covers col tiles 4-at-a-time (one per wave), block = 128 (4 waves)
__global__ __launch_bounds__(128)
void gemm_f16_wmma_kernel(const _Float16* __restrict__ A, const _Float16* __restrict__ Bp,
                          float* __restrict__ C, int K, int Nc) {
  const int NT = Nc >> 4;
  const int KT = K >> 5;
  const int wave = threadIdx.x >> 5;
  const int lane = threadIdx.x & 31;
  const int ct = blockIdx.y * 4 + wave;
  if (ct >= NT) return;                       // wave-uniform: EXEC stays all-1s for WMMA
  const int m0 = blockIdx.x << 4;
  const int hi = lane >> 4;
  const int row = m0 + (lane & 15);
  // A 16x32 f16 layout: lane<16 holds K {0..7,16..23}; lane>=16 holds K {8..15,24..31}
  const _Float16* arow = A + (size_t)row * K + (hi << 3);
  const v16h* bq = (const v16h*)Bp + ((size_t)ct * KT << 5) + lane;
  v8f c = {};
  for (int kt = 0; kt < KT; ++kt) {
    const int kk = kt << 5;
    v8h a0 = *(const v8h*)(arow + kk);
    v8h a1 = *(const v8h*)(arow + kk + 16);
    v16h a;
#pragma unroll
    for (int j = 0; j < 8; ++j) { a[j] = a0[j]; a[j + 8] = a1[j]; }
    v16h b = bq[(size_t)kt << 5];
    if (kt + 1 < KT) __builtin_prefetch(arow + kk + 32, 0, 3);  // global_prefetch_b8
    c = __builtin_amdgcn_wmma_f32_16x16x32_f16(false, a, false, b, (short)0, c, false, false);
  }
  // C 16x16 f32: VGPR r -> M = m0 + r + 8*hi, N = ct*16 + (lane&15)
  const int orow = m0 + (hi << 3);
  const int ocol = (ct << 4) + (lane & 15);
#pragma unroll
  for (int r = 0; r < 8; ++r)
    C[(size_t)(orow + r) * Nc + ocol] = c[r];
}

// ---------------- GAT attention kernels ----------------
__global__ void attn_coef_kernel(const float* __restrict__ xp, const float* __restrict__ aS,
                                 const float* __restrict__ aD, float* __restrict__ als,
                                 float* __restrict__ ald, int H, int Cc, int HC) {
  int t = blockIdx.x * blockDim.x + threadIdx.x;
  if (t >= N_NODES * H) return;
  int n = t / H, h = t % H;
  const float* xr = xp + (size_t)n * HC + h * Cc;
  float s = 0.f, d = 0.f;
  for (int c = 0; c < Cc; ++c) { float v = xr[c]; s += v * aS[h * Cc + c]; d += v * aD[h * Cc + c]; }
  als[t] = s; ald[t] = d;
}

__global__ void edge_max_kernel(const int* __restrict__ ei, const float* __restrict__ als,
                                const float* __restrict__ ald, unsigned* __restrict__ segmax, int H) {
  int t = blockIdx.x * blockDim.x + threadIdx.x;
  if (t >= ET_EDGES * H) return;
  int e = t / H, h = t % H, s, d;
  edge_sd(ei, e, s, d);
  float logit = leaky02(als[s * H + h] + ald[d * H + h]);
  atomicMax(&segmax[d * H + h], f2key(logit));
}

__global__ void edge_sum_kernel(const int* __restrict__ ei, const float* __restrict__ als,
                                const float* __restrict__ ald, const unsigned* __restrict__ segmax,
                                float* __restrict__ segsum, int H) {
  int t = blockIdx.x * blockDim.x + threadIdx.x;
  if (t >= ET_EDGES * H) return;
  int e = t / H, h = t % H, s, d;
  edge_sd(ei, e, s, d);
  float logit = leaky02(als[s * H + h] + ald[d * H + h]);
  float m = key2f(segmax[d * H + h]);
  atomicAdd(&segsum[d * H + h], __expf(logit - m));
}

__global__ void edge_aggr_kernel(const int* __restrict__ ei, const float* __restrict__ als,
                                 const float* __restrict__ ald, const unsigned* __restrict__ segmax,
                                 const float* __restrict__ segsum, const float* __restrict__ xp,
                                 float* __restrict__ outb, int H, int Cc, int HC) {
  int t = blockIdx.x * blockDim.x + threadIdx.x;
  if (t >= ET_EDGES * H) return;
  int e = t / H, h = t % H, s, d;
  edge_sd(ei, e, s, d);
  float logit = leaky02(als[s * H + h] + ald[d * H + h]);
  float m = key2f(segmax[d * H + h]);
  float w = __expf(logit - m) / (segsum[d * H + h] + 1e-16f);
  const float* xs = xp + (size_t)s * HC + h * Cc;
  float* od = outb + (size_t)d * HC + h * Cc;
  for (int c = 0; c < Cc; ++c) atomicAdd(od + c, w * xs[c]);
}

// ---------------- BN / bias / pooling ----------------
__global__ void bias_relu_kernel(float* __restrict__ y, const float* __restrict__ bias, int HC) {
  int i = blockIdx.x * blockDim.x + threadIdx.x;
  if (i >= N_NODES * HC) return;
  float v = y[i] + bias[i % HC];
  y[i] = v > 0.f ? v : 0.f;
}

__global__ void bn_stats_kernel(const float* __restrict__ y, float* __restrict__ stats, int HC) {
  __shared__ float ssum[256], ssq[256];
  int c = blockIdx.x;
  float s = 0.f, q = 0.f;
  for (int n = threadIdx.x; n < N_NODES; n += 256) {
    float v = y[(size_t)n * HC + c];
    s += v; q += v * v;
  }
  ssum[threadIdx.x] = s; ssq[threadIdx.x] = q;
  __syncthreads();
  for (int step = 128; step > 0; step >>= 1) {
    if (threadIdx.x < step) { ssum[threadIdx.x] += ssum[threadIdx.x + step]; ssq[threadIdx.x] += ssq[threadIdx.x + step]; }
    __syncthreads();
  }
  if (threadIdx.x == 0) { stats[c] = ssum[0]; stats[HC + c] = ssq[0]; }
}

__global__ void bn_apply_kernel(float* __restrict__ y, const float* __restrict__ stats,
                                const float* __restrict__ g, const float* __restrict__ b, int HC) {
  int i = blockIdx.x * blockDim.x + threadIdx.x;
  if (i >= N_NODES * HC) return;
  int c = i % HC;
  float m = stats[c] * (1.f / N_NODES);
  float var = stats[HC + c] * (1.f / N_NODES) - m * m;     // biased var
  y[i] = (y[i] - m) * rsqrtf(var + EPS_BN) * g[c] + b[c];
}

__global__ void pool_kernel(const float* __restrict__ h, const int* __restrict__ batch,
                            float* __restrict__ pooled) {
  int i = blockIdx.x * blockDim.x + threadIdx.x;
  if (i >= N_NODES * 96) return;
  int n = i / 96, c = i % 96;
  atomicAdd(&pooled[batch[n] * 96 + c], h[i]);
}

// ---------------- tiny dense head ----------------
__global__ void linear_kernel(const float* __restrict__ in, int ldin, const float* __restrict__ W,
                              const float* __restrict__ bias, float* __restrict__ out, int ldout,
                              const float* __restrict__ res, int ldres, int Gr, int Kin, int Kout,
                              int do_relu) {
  int t = blockIdx.x * blockDim.x + threadIdx.x;
  if (t >= Gr * Kout) return;
  int g = t / Kout, o = t % Kout;
  float acc = bias[o];
  for (int k = 0; k < Kin; ++k) acc += in[g * ldin + k] * W[k * Kout + o];
  if (res) acc += res[g * ldres + o];
  if (do_relu) acc = fmaxf(acc, 0.f);
  out[g * ldout + o] = acc;
}

extern "C" void kernel_launch(void* const* d_in, const int* in_sizes, int n_in,
                              void* d_out, int out_size, void* d_ws, size_t ws_size,
                              hipStream_t stream) {
  (void)in_sizes; (void)n_in; (void)out_size; (void)ws_size;
  const float* x     = (const float*)d_in[0];
  const int*   ei    = (const int*)d_in[1];   // [2,E] row-major: src then dst
  /* d_in[2] = edge_w: unused by reference */
  const int*   batch = (const int*)d_in[3];
  const float* g1_W = (const float*)d_in[4],  *g1_as = (const float*)d_in[5];
  const float* g1_ad = (const float*)d_in[6], *g1_b = (const float*)d_in[7];
  const float* bn1_g = (const float*)d_in[8], *bn1_b = (const float*)d_in[9];
  const float* g2_W = (const float*)d_in[10], *g2_as = (const float*)d_in[11];
  const float* g2_ad = (const float*)d_in[12], *g2_b = (const float*)d_in[13];
  const float* bn2_g = (const float*)d_in[14], *bn2_b = (const float*)d_in[15];
  const float* g3_W = (const float*)d_in[16], *g3_as = (const float*)d_in[17];
  const float* g3_ad = (const float*)d_in[18], *g3_b = (const float*)d_in[19];
  const float* bn3_g = (const float*)d_in[20], *bn3_b = (const float*)d_in[21];
  const float* fc1_W = (const float*)d_in[22], *fc1_b = (const float*)d_in[23];
  const float* r1_W1 = (const float*)d_in[24], *r1_b1 = (const float*)d_in[25];
  const float* r1_W2 = (const float*)d_in[26], *r1_b2 = (const float*)d_in[27];
  const float* r2_W1 = (const float*)d_in[28], *r2_b1 = (const float*)d_in[29];
  const float* r2_W2 = (const float*)d_in[30], *r2_b2 = (const float*)d_in[31];
  const float* r3_W1 = (const float*)d_in[32], *r3_b1 = (const float*)d_in[33];
  const float* r3_W2 = (const float*)d_in[34], *r3_b2 = (const float*)d_in[35];
  const float* fc2_W = (const float*)d_in[36], *fc2_b = (const float*)d_in[37];
  float* outp = (float*)d_out;

  // carve workspace
  char* ws = (char*)d_ws;
  size_t off = 0;
  auto take = [&](size_t bytes) -> char* {
    char* p = ws + off;
    off += (bytes + 255) & ~(size_t)255;
    return p;
  };
  float*    h1     = (float*)take((size_t)N_NODES * 384 * 4);
  float*    h2     = (float*)take((size_t)N_NODES * 384 * 4);
  float*    xp     = (float*)take((size_t)N_NODES * 384 * 4);
  _Float16* a16    = (_Float16*)take((size_t)N_NODES * 384 * 2);
  _Float16* bp16   = (_Float16*)take((size_t)384 * 384 * 2);
  float*    als    = (float*)take((size_t)N_NODES * 3 * 4);
  float*    ald    = (float*)take((size_t)N_NODES * 3 * 4);
  unsigned* segmax = (unsigned*)take((size_t)N_NODES * 3 * 4);
  float*    segsum = (float*)take((size_t)N_NODES * 3 * 4);
  float*    stats  = (float*)take((size_t)2 * 384 * 4);
  float*    pooled = (float*)take((size_t)G_GRAPHS * 96 * 4);
  float*    dense  = (float*)take((size_t)G_GRAPHS * 256 * 4);
  float*    tmp    = (float*)take((size_t)G_GRAPHS * 128 * 4);

  auto gat_layer = [&](const float* hin, float* hout, int Fin, int H, int Cc,
                       const float* W, const float* aS, const float* aD,
                       const float* bias, const float* bng, const float* bnb) {
    const int HC = H * Cc;
    cast_f16_kernel<<<CD(N_NODES * Fin, 256), 256, 0, stream>>>(hin, a16, N_NODES * Fin);
    pack_b_kernel<<<CD(Fin * HC, 256), 256, 0, stream>>>(W, bp16, Fin, HC);
    dim3 gg(N_NODES / 16, CD(HC / 16, 4));
    gemm_f16_wmma_kernel<<<gg, 128, 0, stream>>>(a16, bp16, xp, Fin, HC);
    attn_coef_kernel<<<CD(N_NODES * H, 256), 256, 0, stream>>>(xp, aS, aD, als, ald, H, Cc, HC);
    fill_zero_kernel<<<CD(N_NODES * H, 256), 256, 0, stream>>>((float*)segmax, N_NODES * H);
    fill_zero_kernel<<<CD(N_NODES * H, 256), 256, 0, stream>>>(segsum, N_NODES * H);
    edge_max_kernel<<<CD(ET_EDGES * H, 256), 256, 0, stream>>>(ei, als, ald, segmax, H);
    edge_sum_kernel<<<CD(ET_EDGES * H, 256), 256, 0, stream>>>(ei, als, ald, segmax, segsum, H);
    fill_zero_kernel<<<CD(N_NODES * HC, 256), 256, 0, stream>>>(hout, N_NODES * HC);
    edge_aggr_kernel<<<CD(ET_EDGES * H, 256), 256, 0, stream>>>(ei, als, ald, segmax, segsum, xp, hout, H, Cc, HC);
    bias_relu_kernel<<<CD(N_NODES * HC, 256), 256, 0, stream>>>(hout, bias, HC);
    bn_stats_kernel<<<HC, 256, 0, stream>>>(hout, stats, HC);
    bn_apply_kernel<<<CD(N_NODES * HC, 256), 256, 0, stream>>>(hout, stats, bng, bnb, HC);
  };

  gat_layer(x,  h1, 128, 3, 128, g1_W, g1_as, g1_ad, g1_b, bn1_g, bn1_b);
  gat_layer(h1, h2, 384, 3, 64,  g2_W, g2_as, g2_ad, g2_b, bn2_g, bn2_b);
  gat_layer(h2, h1, 192, 3, 32,  g3_W, g3_as, g3_ad, g3_b, bn3_g, bn3_b);

  // global_add_pool -> [G,96]
  fill_zero_kernel<<<CD(G_GRAPHS * 96, 256), 256, 0, stream>>>(pooled, G_GRAPHS * 96);
  pool_kernel<<<CD(N_NODES * 96, 256), 256, 0, stream>>>(h1, batch, pooled);

  // MLP head; dense = [cr | x1 | x2 | x3] as columns of [G,256]
  linear_kernel<<<CD(64 * 64, 256), 256, 0, stream>>>(pooled, 96, fc1_W, fc1_b, dense + 0, 256, nullptr, 0, 64, 96, 64, 1);
  // res block 1
  linear_kernel<<<CD(64 * 128, 256), 256, 0, stream>>>(dense + 0, 256, r1_W1, r1_b1, tmp, 128, nullptr, 0, 64, 64, 128, 1);
  linear_kernel<<<CD(64 * 64, 256), 256, 0, stream>>>(tmp, 128, r1_W2, r1_b2, dense + 64, 256, dense + 0, 256, 64, 128, 64, 1);
  // res block 2
  linear_kernel<<<CD(64 * 128, 256), 256, 0, stream>>>(dense + 64, 256, r2_W1, r2_b1, tmp, 128, nullptr, 0, 64, 64, 128, 1);
  linear_kernel<<<CD(64 * 64, 256), 256, 0, stream>>>(tmp, 128, r2_W2, r2_b2, dense + 128, 256, dense + 64, 256, 64, 128, 64, 1);
  // res block 3
  linear_kernel<<<CD(64 * 128, 256), 256, 0, stream>>>(dense + 128, 256, r3_W1, r3_b1, tmp, 128, nullptr, 0, 64, 64, 128, 1);
  linear_kernel<<<CD(64 * 64, 256), 256, 0, stream>>>(tmp, 128, r3_W2, r3_b2, dense + 192, 256, dense + 128, 256, 64, 128, 64, 1);
  // fc2
  linear_kernel<<<CD(64 * 32, 256), 256, 0, stream>>>(dense, 256, fc2_W, fc2_b, outp, 32, nullptr, 0, 64, 256, 32, 1);
}
